// Net_Deform_19988777795627
// MI455X (gfx1250) — matrix operand order repaired and checked by reading, not compiled, not run on hardware
//
#include <hip/hip_runtime.h>
#include <cmath>

// ---------------------------------------------------------------------------
// Types for CDNA5 WMMA (wave32): A/B = 16 x f16 per lane, C/D = 8 x f32.
// ---------------------------------------------------------------------------
typedef __attribute__((ext_vector_type(16))) _Float16 v16h;
typedef __attribute__((ext_vector_type(8)))  float    v8f;

union HVec { v16h v; _Float16 h[16]; };

#define BATCH 2048

// ---------------------------------------------------------------------------
// Bilinear sample matching the reference (floor / ceil corners).
// ---------------------------------------------------------------------------
__device__ __forceinline__ float bilinear(const float* __restrict__ img,
                                          int W, float cy, float cx) {
    float y0f = floorf(cy), x0f = floorf(cx);
    int y0 = (int)y0f,        x0 = (int)x0f;
    int y1 = (int)ceilf(cy),  x1 = (int)ceilf(cx);
    float v_lt = img[y0 * W + x0];
    float v_rt = img[y0 * W + x1];
    float v_lb = img[y1 * W + x0];
    float v_rb = img[y1 * W + x1];
    float tx = cx - x0f, ty = cy - y0f;
    float v_t = v_lt + (v_rt - v_lt) * tx;
    float v_b = v_lb + (v_rb - v_lb) * tx;
    return v_t + (v_b - v_t) * ty;
}

// ---------------------------------------------------------------------------
// Kernel 1: block-1 deformable sampling on x (B,1,28,28).
// Offsets come from a 3x3 conv (1->2ch) evaluated at the *raw-reshape*
// scrambled flat position, exactly like the repo.
// ---------------------------------------------------------------------------
__global__ void k_deform1(const float* __restrict__ x,
                          const float* __restrict__ ow,   // (2,1,3,3)
                          float* __restrict__ xd) {
    const int HW = 28 * 28;
    int idx = blockIdx.x * blockDim.x + threadIdx.x;
    if (idx >= BATCH * HW) return;
    int b  = idx / HW;
    int hw = idx % HW;
    int h = hw / 28, w = hw % 28;
    const float* xb = x + b * HW;

    float off[2];
    int f0 = hw * 2;
#pragma unroll
    for (int k = 0; k < 2; ++k) {
        int f  = f0 + k;                 // flat into (2,28,28) block
        int ch = f / HW;
        int r  = f % HW;
        int hh = r / 28, ww = r % 28;
        float acc = 0.f;
#pragma unroll
        for (int i = 0; i < 3; ++i)
#pragma unroll
            for (int j = 0; j < 3; ++j) {
                int yy = hh + i - 1, xx = ww + j - 1;
                if (yy >= 0 && yy < 28 && xx >= 0 && xx < 28)
                    acc += xb[yy * 28 + xx] * ow[ch * 9 + i * 3 + j];
            }
        off[k] = acc;
    }
    float cy = fminf(fmaxf(off[0] + (float)h, 0.f), 27.f);
    float cx = fminf(fmaxf(off[1] + (float)w, 0.f), 27.f);
    xd[idx] = bilinear(xb, 28, cy, cx);
}

// ---------------------------------------------------------------------------
// Kernel 2: conv1 5x5 pad2 (1->16) + ReLU + 2x2 maxpool -> h1 (B,16,14,14).
// Cin=1, so VALU is the right tool (25 MACs per conv output).
// ---------------------------------------------------------------------------
__global__ void k_conv1_pool(const float* __restrict__ xd,
                             const float* __restrict__ w,    // (16,1,5,5)
                             const float* __restrict__ bias, // (16)
                             float* __restrict__ h1) {
    int idx = blockIdx.x * blockDim.x + threadIdx.x;
    const int TOT = BATCH * 16 * 14 * 14;
    if (idx >= TOT) return;
    int pw = idx % 14;
    int ph = (idx / 14) % 14;
    int c  = (idx / 196) % 16;
    int b  = idx / (196 * 16);

    const float* xb = xd + b * 784;
    const float* wc = w + c * 25;
    float wreg[25];
#pragma unroll
    for (int i = 0; i < 25; ++i) wreg[i] = wc[i];
    float bb = bias[c];

    float best = -INFINITY;
#pragma unroll
    for (int dy = 0; dy < 2; ++dy)
#pragma unroll
        for (int dx = 0; dx < 2; ++dx) {
            int oh = 2 * ph + dy, owp = 2 * pw + dx;
            float acc = bb;
#pragma unroll
            for (int i = 0; i < 5; ++i) {
                int yy = oh + i - 2;
                if (yy < 0 || yy >= 28) continue;
#pragma unroll
                for (int j = 0; j < 5; ++j) {
                    int xx = owp + j - 2;
                    if (xx >= 0 && xx < 28)
                        acc += xb[yy * 28 + xx] * wreg[i * 5 + j];
                }
            }
            best = fmaxf(best, fmaxf(acc, 0.f));
        }
    h1[idx] = best;
}

// ---------------------------------------------------------------------------
// Implicit-GEMM conv via WMMA.  in:(B,CIN,14,14) -> out:(B,32,14,14).
//   M = B*196 rows (output pixels), N = 32 (Cout), K = CIN*KS*KS.
// 128 threads = 4 waves; each wave owns 16 M-rows and both 16-wide N tiles.
// Weights (B-matrix, K x 32) are staged to LDS ONCE (<=26 KB of the 320 KB
// WGP LDS).  The K loop only re-stages the 4 KB im2col A tile, and while
// staging chunk kc it prefetches chunk kc+1's sources (global_prefetch_b8).
// Fragments are gathered in contiguous k order (k = 16*half + e) for A and B, then
// v_wmma_f32_16x16x32_f16 accumulates in f32.
// M = 2048*196 = 401408 is divisible by 64 -> no tail handling.
// ---------------------------------------------------------------------------
template <int CIN, int KS, int PAD, int KTOT, bool RELU_BIAS>
__global__ void __launch_bounds__(128)
k_conv_wmma(const float* __restrict__ in,
            const float* __restrict__ wgt,   // (32, CIN, KS, KS)
            const float* __restrict__ bias,  // (32) or nullptr
            float* __restrict__ out) {
    constexpr int KCH  = (KTOT + 31) / 32;
    constexpr int KPAD = KCH * 32;
    constexpr int HW   = 14 * 14;

    __shared__ _Float16 sA[64 * 32];      // 64 M-rows x 32 K (per chunk)
    __shared__ _Float16 sB[KPAD * 32];    // full K x 32 N (weights)

    const int tid   = threadIdx.x;
    const int wv    = tid >> 5;
    const int lane  = tid & 31;
    const int half  = lane >> 4;
    const int ln16  = lane & 15;
    const int pBase = blockIdx.x * 64;

    // ---- stage full B (weights) once ----
    for (int i = tid; i < KPAD * 32; i += 128) {
        int kk = i >> 5;
        int co = i & 31;
        float v = (kk < KTOT) ? wgt[co * KTOT + kk] : 0.f;
        sB[kk * 32 + co] = (_Float16)v;
    }

    v8f acc0 = {};
    v8f acc1 = {};

    for (int kc = 0; kc < KCH; ++kc) {
        __syncthreads();   // (also covers initial sB visibility)
        // ---- stage A chunk (im2col, f32 -> f16) + prefetch next chunk ----
        for (int i = tid; i < 64 * 32; i += 128) {
            int row = i >> 5;
            int kk  = i & 31;
            int p   = pBase + row;
            int b   = p / HW, hw = p % HW;
            int h   = hw / 14, w = hw % 14;

            int kg = kc * 32 + kk;
            float v = 0.f;
            if (kg < KTOT) {
                int ci = kg / (KS * KS), t = kg % (KS * KS);
                int kh = t / KS, kw = t % KS;
                int y  = h + kh - PAD, x = w + kw - PAD;
                if (y >= 0 && y < 14 && x >= 0 && x < 14)
                    v = in[((b * CIN + ci) * 14 + y) * 14 + x];
            }
            sA[i] = (_Float16)v;

            // speculative prefetch of next chunk's source cacheline
            int kg2 = (kc + 1) * 32 + kk;
            if (kg2 < KTOT) {
                int ci2 = kg2 / (KS * KS);
                __builtin_prefetch(&in[((b * CIN + ci2) * 14 + h) * 14 + w], 0, 0);
            }
        }
        __syncthreads();

        // ---- gather fragments, contiguous k = 16*half + e for A and B ----
        HVec a;
#pragma unroll
        for (int e = 0; e < 16; ++e) {
            int kk = half * 16 + e;                       // A 16x32 layout
            a.h[e] = sA[(wv * 16 + ln16) * 32 + kk];
        }
        HVec b0, b1;
#pragma unroll
        for (int e = 0; e < 16; ++e) {
            int kk = kc * 32 + half * 16 + e;             // B 32x16 layout
            b0.h[e] = sB[kk * 32 + ln16];
            b1.h[e] = sB[kk * 32 + 16 + ln16];
        }
        acc0 = __builtin_amdgcn_wmma_f32_16x16x32_f16(
            false, a.v, false, b0.v, (short)0, acc0, false, false);
        acc1 = __builtin_amdgcn_wmma_f32_16x16x32_f16(
            false, a.v, false, b1.v, (short)0, acc1, false, false);
    }

    // ---- store: D layout -> VGPR r holds M = r + 8*half, N = lane%16 ----
#pragma unroll
    for (int r = 0; r < 8; ++r) {
        int m  = r + 8 * half;
        int p  = pBase + wv * 16 + m;
        int b  = p / HW, hw = p % HW;
        float v0 = acc0[r];
        float v1 = acc1[r];
        int co0 = ln16, co1 = 16 + ln16;
        if constexpr (RELU_BIAS) {
            v0 = fmaxf(v0 + bias[co0], 0.f);
            v1 = fmaxf(v1 + bias[co1], 0.f);
        }
        out[(b * 32 + co0) * HW + hw] = v0;
        out[(b * 32 + co1) * HW + hw] = v1;
    }
}

// ---------------------------------------------------------------------------
// Kernel 4: block-2 deformable sampling.  off2 (B,32,14,14) is materialized,
// so the repo's raw reshape is a direct flat index into each batch's block.
// ---------------------------------------------------------------------------
__global__ void k_deform2(const float* __restrict__ h1,
                          const float* __restrict__ off2,
                          float* __restrict__ h1d) {
    const int HW = 14 * 14;
    int idx = blockIdx.x * blockDim.x + threadIdx.x;
    const int TOT = BATCH * 16 * HW;
    if (idx >= TOT) return;
    int hw = idx % HW;
    int c  = (idx / HW) % 16;
    int b  = idx / (HW * 16);
    int h = hw / 14, w = hw % 14;

    const float* ob = off2 + (size_t)b * 32 * HW;
    int f0 = c * 2 * HW + hw * 2;
    float oy = ob[f0];
    float ox = ob[f0 + 1];
    float cy = fminf(fmaxf(oy + (float)h, 0.f), 13.f);
    float cx = fminf(fmaxf(ox + (float)w, 0.f), 13.f);
    h1d[idx] = bilinear(h1 + (size_t)(b * 16 + c) * HW, 14, cy, cx);
}

// ---------------------------------------------------------------------------
// Kernel 6: 2x2 maxpool (B,32,14,14) -> (B,32,7,7) (== FC flat layout).
// ---------------------------------------------------------------------------
__global__ void k_pool2(const float* __restrict__ src, float* __restrict__ dst) {
    int idx = blockIdx.x * blockDim.x + threadIdx.x;
    const int TOT = BATCH * 32 * 49;
    if (idx >= TOT) return;
    int pw = idx % 7;
    int ph = (idx / 7) % 7;
    int c  = (idx / 49) % 32;
    int b  = idx / (49 * 32);
    const float* s = src + (size_t)(b * 32 + c) * 196;
    int y = 2 * ph, x = 2 * pw;
    float m = fmaxf(fmaxf(s[y * 14 + x], s[y * 14 + x + 1]),
                    fmaxf(s[(y + 1) * 14 + x], s[(y + 1) * 14 + x + 1]));
    dst[idx] = m;
}

// ---------------------------------------------------------------------------
// Kernel 7: FC via WMMA.  M=2048, K=1568 (exactly 49 x 32), N=10 padded to 16.
// fc_w (10x1568 -> padded 16-wide, ~50 KB f16) staged to LDS once.
// out[m,n] = sum_k flat[m,k] * fc_w[n,k] + fc_b[n]
// ---------------------------------------------------------------------------
__global__ void __launch_bounds__(128)
k_fc_wmma(const float* __restrict__ flat,  // (B,1568)
          const float* __restrict__ fw,    // (10,1568)
          const float* __restrict__ fb,    // (10)
          float* __restrict__ out) {       // (B,10)
    __shared__ _Float16 sA[64 * 32];
    __shared__ _Float16 sB[1568 * 16];

    const int tid   = threadIdx.x;
    const int wv    = tid >> 5;
    const int lane  = tid & 31;
    const int half  = lane >> 4;
    const int ln16  = lane & 15;
    const int mBase = blockIdx.x * 64;

    // ---- stage full B (fc weights) once ----
    for (int i = tid; i < 1568 * 16; i += 128) {
        int kk = i >> 4;
        int n  = i & 15;
        float v = (n < 10) ? fw[(size_t)n * 1568 + kk] : 0.f;
        sB[kk * 16 + n] = (_Float16)v;
    }

    v8f acc = {};

    for (int kc = 0; kc < 49; ++kc) {
        __syncthreads();
        for (int i = tid; i < 64 * 32; i += 128) {
            int row = i >> 5;
            int kk  = i & 31;
            const float* src = &flat[(size_t)(mBase + row) * 1568 + kc * 32 + kk];
            sA[i] = (_Float16)(*src);
            if (kc + 1 < 49) __builtin_prefetch(src + 32, 0, 0);
        }
        __syncthreads();

        HVec a, bfr;
#pragma unroll
        for (int e = 0; e < 16; ++e) {
            int kk = half * 16 + e;
            a.h[e] = sA[(wv * 16 + ln16) * 32 + kk];
        }
#pragma unroll
        for (int e = 0; e < 16; ++e) {
            int kk = kc * 32 + half * 16 + e;
            bfr.h[e] = sB[kk * 16 + ln16];
        }
        acc = __builtin_amdgcn_wmma_f32_16x16x32_f16(
            false, a.v, false, bfr.v, (short)0, acc, false, false);
    }

#pragma unroll
    for (int r = 0; r < 8; ++r) {
        int m = mBase + wv * 16 + r + 8 * half;
        int n = ln16;
        if (n < 10) out[m * 10 + n] = acc[r] + fb[n];
    }
}

// ---------------------------------------------------------------------------
// Host-side orchestration (all on `stream`, graph-capture safe).
// ---------------------------------------------------------------------------
extern "C" void kernel_launch(void* const* d_in, const int* in_sizes, int n_in,
                              void* d_out, int out_size, void* d_ws, size_t ws_size,
                              hipStream_t stream) {
    (void)in_sizes; (void)n_in; (void)out_size; (void)ws_size;

    const float* x       = (const float*)d_in[0];
    const float* off1_w  = (const float*)d_in[1];
    const float* conv1_w = (const float*)d_in[2];
    const float* conv1_b = (const float*)d_in[3];
    const float* off2_w  = (const float*)d_in[4];
    const float* conv2_w = (const float*)d_in[5];
    const float* conv2_b = (const float*)d_in[6];
    const float* fc_w    = (const float*)d_in[7];
    const float* fc_b    = (const float*)d_in[8];
    float* out = (float*)d_out;

    // Workspace layout (bytes), with reuse of dead buffers:
    const size_t SZ_XD   = (size_t)BATCH * 784 * sizeof(float);        // 6.4 MB
    const size_t SZ_H1   = (size_t)BATCH * 16 * 196 * sizeof(float);   // 25.7 MB
    const size_t SZ_OFF2 = (size_t)BATCH * 32 * 196 * sizeof(float);   // 51.4 MB

    char* ws = (char*)d_ws;
    float* xd     = (float*)(ws);
    float* h1     = (float*)(ws + SZ_XD);
    float* off2   = (float*)(ws + SZ_XD + SZ_H1);
    float* h1d    = (float*)(ws + SZ_XD + SZ_H1 + SZ_OFF2);
    float* conv2o = off2;   // off2 dead after k_deform2
    float* pooled = h1;     // h1 dead after k_deform2

    // 1) deformable sampling of x
    {
        int tot = BATCH * 784;
        k_deform1<<<(tot + 255) / 256, 256, 0, stream>>>(x, off1_w, xd);
    }
    // 2) conv1 + relu + pool -> h1 (B,16,14,14)
    {
        int tot = BATCH * 16 * 196;
        k_conv1_pool<<<(tot + 255) / 256, 256, 0, stream>>>(xd, conv1_w, conv1_b, h1);
    }
    // 3) offset conv2 (16->32, 3x3 pad1) via WMMA -> off2 (B,32,14,14)
    {
        int blocks = (BATCH * 196) / 64;   // 6272, exact
        k_conv_wmma<16, 3, 1, 144, false><<<blocks, 128, 0, stream>>>(
            h1, off2_w, nullptr, off2);
    }
    // 4) deformable sampling of h1 -> h1d
    {
        int tot = BATCH * 16 * 196;
        k_deform2<<<(tot + 255) / 256, 256, 0, stream>>>(h1, off2, h1d);
    }
    // 5) conv2 (16->32, 5x5 pad2) + bias + relu via WMMA -> conv2o (B,32,14,14)
    {
        int blocks = (BATCH * 196) / 64;
        k_conv_wmma<16, 5, 2, 400, true><<<blocks, 128, 0, stream>>>(
            h1d, conv2_w, conv2_b, conv2o);
    }
    // 6) maxpool -> pooled (B,32,7,7) == FC input (B,1568)
    {
        int tot = BATCH * 32 * 49;
        k_pool2<<<(tot + 255) / 256, 256, 0, stream>>>(conv2o, pooled);
    }
    // 7) FC via WMMA -> out (B,10)
    {
        k_fc_wmma<<<BATCH / 64, 128, 0, stream>>>(pooled, fc_w, fc_b, out);
    }
}
